// RIPointTransformerLayer_4234837754417
// MI455X (gfx1250) — matrix operand order repaired and checked by reading, not compiled, not run on hardware
//
#include <hip/hip_runtime.h>
#include <math.h>

// MI455X / gfx1250, wave32.
//  - project x once (XQ/XK/XV = x@W+b) with f16 WMMA; gathers hit 192MB L2.
//  - fuse PPF -> relu(ppf@Wp1+b)@Wp2+b -> attention per 8-point tile; the
//    640k x 128 x 128 embedding GEMM (~21 GFLOP, dominant) runs on
//    v_wmma_f32_16x16x32_f16 with e kept in LDS (never HBM).
//  - gathered XV rows are DMA'd into LDS with global_load_async_to_lds_b128,
//    overlapping the WMMA GEMM; s_wait_asynccnt fences before use.
//  - weights stored TRANSPOSED in LDS so every WMMA B fragment is two
//    contiguous ds_load_b128 (no scalar u16 assembly); padded strides keep
//    LDS banks conflict-free.

typedef __attribute__((ext_vector_type(16))) _Float16 v16h;
typedef __attribute__((ext_vector_type(8)))  _Float16 v8h;
typedef __attribute__((ext_vector_type(4)))  _Float16 v4h;
typedef __attribute__((ext_vector_type(8)))  float    v8f;
typedef __attribute__((ext_vector_type(4)))  float    v4f;

#define N_PTS   40000
#define K_NB    16
#define DIM     128
#define HEADS   4
#define DHEAD   32
#define LDH     136        // f16 LDS row stride (272B: 16B-aligned, banks spread)
#define LDF     132        // f32 LDS row stride (528B: 16B-aligned, banks spread)
#define TILE_P  8          // points per workgroup in the fused kernel
#define ROWS    (TILE_P * K_NB)   // 128 (point,neighbor) rows per workgroup

// ---------- wave32 WMMA fragment loaders (contiguous k = 16*half + s, same in A and B) ------

// A (16x32 f16, MxK): lanes 0-15 row M=lane hold K=0..15; lanes 16-31
// same rows, K=16..31.  Two ds_load_b128 per lane.
__device__ __forceinline__ v16h load_a_frag(const _Float16* s, int ld,
                                            int mbase, int kbase, int lane) {
    const int l  = lane & 15;
    const int hi = lane >> 4;
    const v8h* p = (const v8h*)(s + (mbase + l) * ld + kbase);
    const v8h a0 = p[2 * hi];      // halves hi*16 .. hi*16+7
    const v8h a1 = p[2 * hi + 1];  // halves hi*16+8 .. hi*16+15
    return __builtin_shufflevector(a0, a1, 0, 1, 2, 3, 4, 5, 6, 7,
                                   8, 9, 10, 11, 12, 13, 14, 15);
}

// B (32x16 f16, KxN) from a TRANSPOSED weight tile sWt[n][k]: lane = column
// N=nbase+l; lanes 0-15 hold K=kbase..kbase+15, lanes 16-31 K=kbase+16..+31.
// Contiguous in k -> two ds_load_b128 per lane.
__device__ __forceinline__ v16h load_bT_frag(const _Float16* sWt, int ld,
                                             int kbase, int nbase, int lane) {
    const int l  = lane & 15;
    const int hi = lane >> 4;
    const v8h* p = (const v8h*)(sWt + (nbase + l) * ld + kbase + hi * 16);
    const v8h b0 = p[0];
    const v8h b1 = p[1];
    return __builtin_shufflevector(b0, b1, 0, 1, 2, 3, 4, 5, 6, 7,
                                   8, 9, 10, 11, 12, 13, 14, 15);
}

__device__ __forceinline__ v8f v8f_zero() {
    v8f z;
#pragma unroll
    for (int i = 0; i < 8; ++i) z[i] = 0.0f;
    return z;
}

__device__ __forceinline__ float angle3(float ax, float ay, float az,
                                        float bx, float by, float bz) {
    float cx = ay * bz - az * by;
    float cy = az * bx - ax * bz;
    float cz = ax * by - ay * bx;
    float cr = sqrtf(cx * cx + cy * cy + cz * cz);
    float dt = ax * bx + ay * by + az * bz;
    return atan2f(cr, dt);
}

// ---------------- Y[nrows,128] = X[nrows,128] @ W[128,128] + b ---------------
__global__ __launch_bounds__(256)
void gemm128_bias_kernel(const float* __restrict__ X, const float* __restrict__ W,
                         const float* __restrict__ bias, float* __restrict__ Y,
                         int nrows) {
    __shared__ alignas(16) _Float16 sX[DIM * LDH];
    __shared__ alignas(16) _Float16 sWt[DIM * LDH];   // transposed: sWt[n][k]
    __shared__ float sB[DIM];

    const int t    = threadIdx.x;
    const int lane = t & 31;
    const int wid  = t >> 5;
    const int n0   = blockIdx.x * DIM;

    // stage X tile f32 -> f16, vectorized (global b128 load, LDS b64 store)
    for (int i = t; i < DIM * DIM / 4; i += 256) {
        const int r  = i >> 5;
        const int c4 = (i & 31) * 4;
        v4f xv;
        if (n0 + r < nrows) xv = *(const v4f*)(X + (size_t)(n0 + r) * DIM + c4);
        else { xv[0] = xv[1] = xv[2] = xv[3] = 0.0f; }
        v4h h;
#pragma unroll
        for (int j = 0; j < 4; ++j) h[j] = (_Float16)xv[j];
        *(v4h*)(sX + r * LDH + c4) = h;
    }
    // stage W transposed (so B fragments are contiguous ds_load_b128)
    for (int i = t; i < DIM * DIM; i += 256) {
        const int nn = i >> 7, kk = i & 127;
        sWt[nn * LDH + kk] = (_Float16)W[kk * DIM + nn];
    }
    if (t < DIM) sB[t] = bias[t];
    __syncthreads();

    const int mbase = wid * 16;
    v8f acc[8];
#pragma unroll
    for (int nt = 0; nt < 8; ++nt) acc[nt] = v8f_zero();

#pragma unroll
    for (int kb = 0; kb < 4; ++kb) {
        const v16h a = load_a_frag(sX, LDH, mbase, kb * 32, lane);
#pragma unroll
        for (int nt = 0; nt < 8; ++nt) {
            const v16h b = load_bT_frag(sWt, LDH, kb * 32, nt * 16, lane);
            acc[nt] = __builtin_amdgcn_wmma_f32_16x16x32_f16(
                false, a, false, b, (short)0, acc[nt], false, false);
        }
    }

    // D layout: lanes 0-15 -> M=r, lanes 16-31 -> M=8+r; N = lane&15.
    const int l  = lane & 15;
    const int hi = lane >> 4;
#pragma unroll
    for (int nt = 0; nt < 8; ++nt) {
        const int   col = nt * 16 + l;
        const float bv  = sB[col];
#pragma unroll
        for (int r = 0; r < 8; ++r) {
            const int row = n0 + mbase + hi * 8 + r;
            if (row < nrows) Y[(size_t)row * DIM + col] = acc[nt][r] + bv;
        }
    }
}

// -------- fused: PPF -> relu(ppf@Wp1+b)@Wp2+b (WMMA) -> attention -----------
__global__ __launch_bounds__(256)
void ppf_attn_kernel(const float* __restrict__ P,   const float* __restrict__ Nrm,
                     const int*   __restrict__ gidx,
                     const float* __restrict__ Wp1, const float* __restrict__ bp1,
                     const float* __restrict__ Wp2, const float* __restrict__ bp2,
                     const float* __restrict__ XQ,  const float* __restrict__ XK,
                     const float* __restrict__ XV,  float* __restrict__ OUTV) {
    __shared__ alignas(16) _Float16 sWt[DIM * LDH];    // Wp2 transposed, 34KB
    __shared__ alignas(16) _Float16 sH1[ROWS * LDH];   // 34KB
    __shared__ alignas(16) float    sE[ROWS * LDF];    // 66KB, e stays on-chip
    __shared__ alignas(16) float    sXv[ROWS * LDF];   // 66KB, async-gathered XV
    __shared__ float sWp1[4 * DIM];
    __shared__ float sBp1[DIM];
    __shared__ float sBp2[DIM];
    __shared__ alignas(16) float sXq[TILE_P * DIM];
    __shared__ float sPpf[ROWS * 4];
    __shared__ int   sIdx[ROWS];
    __shared__ float sAttn[TILE_P * HEADS * K_NB];

    const int t    = threadIdx.x;
    const int lane = t & 31;
    const int wid  = t >> 5;
    const int n0   = blockIdx.x * TILE_P;

    // stage Wp2 transposed, Wp1/biases, q rows
    for (int i = t; i < DIM * DIM; i += 256) {
        const int nn = i >> 7, kk = i & 127;
        sWt[nn * LDH + kk] = (_Float16)Wp2[kk * DIM + nn];
    }
    for (int i = t; i < 4 * DIM; i += 256) sWp1[i] = Wp1[i];
    if (t < DIM) { sBp1[t] = bp1[t]; sBp2[t] = bp2[t]; }
    {   // 8x128 floats = 256 v4f chunks, one per thread
        const int m = t >> 5, c4 = (t & 31) * 4;
        *(v4f*)(sXq + m * DIM + c4) = *(const v4f*)(XQ + (size_t)(n0 + m) * DIM + c4);
    }

    // PPF features, one (point,neighbor) row per thread
    if (t < ROWS) {
        const int m   = t >> 4;
        const int kk  = t & 15;
        const int n   = n0 + m;
        const int idx = gidx[(size_t)n * K_NB + kk];
        sIdx[t] = idx;
        __builtin_prefetch((const void*)(XK + (size_t)idx * DIM), 0, 1);

        const float pcx = P[n * 3 + 0],   pcy = P[n * 3 + 1],   pcz = P[n * 3 + 2];
        const float ncx = Nrm[n * 3 + 0], ncy = Nrm[n * 3 + 1], ncz = Nrm[n * 3 + 2];
        const float prx = P[idx * 3 + 0], pry = P[idx * 3 + 1], prz = P[idx * 3 + 2];
        const float nrx = Nrm[idx * 3 + 0], nry = Nrm[idx * 3 + 1], nrz = Nrm[idx * 3 + 2];
        const float dx = prx - pcx, dy = pry - pcy, dz = prz - pcz;
        sPpf[t * 4 + 0] = angle3(ncx, ncy, ncz, dx, dy, dz);
        sPpf[t * 4 + 1] = angle3(nrx, nry, nrz, dx, dy, dz);
        sPpf[t * 4 + 2] = angle3(ncx, ncy, ncz, nrx, nry, nrz);
        sPpf[t * 4 + 3] = sqrtf(dx * dx + dy * dy + dz * dz);
    }
    __syncthreads();

    // async DMA: gather the 128 XV rows into LDS (overlaps h1 + WMMA below).
    // 4096 x 16B chunks; each wave instruction moves 512B (one row).
    for (int i = t; i < ROWS * 32; i += 256) {
        const int row = i >> 5;
        const int ch  = i & 31;
        const float*  src = XV + (size_t)sIdx[row] * DIM + ch * 4;
        const unsigned dst =
            (unsigned)(size_t)(const void*)(sXv + row * LDF + ch * 4);
        asm volatile("global_load_async_to_lds_b128 %0, %1, off"
                     :: "v"(dst), "v"(src) : "memory");
    }

    // h1 = relu(ppf @ Wp1 + bp1): K=4 on VALU; 256 threads, half-row each
    {
        const int row = t >> 1;
        const int c0  = (t & 1) * 64;
        const float f0 = sPpf[row * 4 + 0], f1 = sPpf[row * 4 + 1];
        const float f2 = sPpf[row * 4 + 2], f3 = sPpf[row * 4 + 3];
#pragma unroll 4
        for (int c = c0; c < c0 + 64; ++c) {
            float h = sBp1[c] + f0 * sWp1[0 * DIM + c] + f1 * sWp1[1 * DIM + c]
                              + f2 * sWp1[2 * DIM + c] + f3 * sWp1[3 * DIM + c];
            sH1[row * LDH + c] = (_Float16)fmaxf(h, 0.0f);
        }
    }
    __syncthreads();

    // E[128,128] = h1 @ Wp2 + bp2 on the matrix pipe (dominant FLOPs)
    {
        const int mbase = wid * 16;
        v8f acc[8];
#pragma unroll
        for (int nt = 0; nt < 8; ++nt) acc[nt] = v8f_zero();
#pragma unroll
        for (int kb = 0; kb < 4; ++kb) {
            const v16h a = load_a_frag(sH1, LDH, mbase, kb * 32, lane);
#pragma unroll
            for (int nt = 0; nt < 8; ++nt) {
                const v16h b = load_bT_frag(sWt, LDH, kb * 32, nt * 16, lane);
                acc[nt] = __builtin_amdgcn_wmma_f32_16x16x32_f16(
                    false, a, false, b, (short)0, acc[nt], false, false);
            }
        }
        const int l  = lane & 15;
        const int hi = lane >> 4;
#pragma unroll
        for (int nt = 0; nt < 8; ++nt) {
            const int   col = nt * 16 + l;
            const float bv  = sBp2[col];
#pragma unroll
            for (int r = 0; r < 8; ++r)
                sE[(mbase + hi * 8 + r) * LDF + col] = acc[nt][r] + bv;
        }
    }
    __syncthreads();

    // attention logits + softmax: wave w -> point m=w; lanes 0-15 / 16-31 are
    // two width-16 groups, one head each per iteration (wave32-native shuffles)
    {
        const int m   = wid;
        const int kk  = lane & 15;
        const int hh  = lane >> 4;
        const int row = m * K_NB + kk;
        const int idx = sIdx[row];
        const float* eRow = sE  + row * LDF;
        const float* qRow = sXq + m * DIM;
        const float* kRow = XK  + (size_t)idx * DIM;   // L2-resident gather
#pragma unroll
        for (int it = 0; it < 2; ++it) {
            const int h   = it * 2 + hh;
            const int off = h * DHEAD;
            float dot = 0.0f;
#pragma unroll
            for (int d = 0; d < DHEAD; ++d)
                dot += qRow[off + d] * (kRow[off + d] + eRow[off + d]);
            const float logit = dot * 0.17677669529663688f;   // 1/sqrt(32)
            float mx = logit;
#pragma unroll
            for (int o = 8; o >= 1; o >>= 1) mx = fmaxf(mx, __shfl_xor(mx, o, 16));
            const float ex = __expf(logit - mx);
            float sm = ex;
#pragma unroll
            for (int o = 8; o >= 1; o >>= 1) sm += __shfl_xor(sm, o, 16);
            sAttn[(m * HEADS + h) * K_NB + kk] = ex / sm;
        }
    }

    // fence the async XV gather, then everyone syncs before consuming sXv
    asm volatile("s_wait_asynccnt 0x0" ::: "memory");
    __syncthreads();

    // out[m,col] = sum_k attn[m,h(col),k] * (XV_lds + e); 1024 outputs, 4/thread
#pragma unroll
    for (int j = 0; j < 4; ++j) {
        const int lin = t + j * 256;
        const int m   = lin >> 7;
        const int col = lin & 127;
        const int h   = col >> 5;
        const float* aw = sAttn + (m * HEADS + h) * K_NB;
        float acc = 0.0f;
#pragma unroll
        for (int kk = 0; kk < K_NB; ++kk) {
            const int row = m * K_NB + kk;
            acc += aw[kk] * (sXv[row * LDF + col] + sE[row * LDF + col]);
        }
        OUTV[(size_t)(n0 + m) * DIM + col] = acc;
    }
}

extern "C" void kernel_launch(void* const* d_in, const int* in_sizes, int n_in,
                              void* d_out, int out_size, void* d_ws, size_t ws_size,
                              hipStream_t stream) {
    const float* p    = (const float*)d_in[0];
    const float* nrm  = (const float*)d_in[1];
    const float* x    = (const float*)d_in[2];
    const int*   gidx = (const int*)  d_in[3];
    const float* Wq   = (const float*)d_in[4];
    const float* bq   = (const float*)d_in[5];
    const float* Wk   = (const float*)d_in[6];
    const float* bk   = (const float*)d_in[7];
    const float* Wv   = (const float*)d_in[8];
    const float* bv   = (const float*)d_in[9];
    const float* Wp1  = (const float*)d_in[10];
    const float* bp1  = (const float*)d_in[11];
    const float* Wp2  = (const float*)d_in[12];
    const float* bp2  = (const float*)d_in[13];
    const float* Wo   = (const float*)d_in[14];
    const float* bo   = (const float*)d_in[15];
    float* out = (float*)d_out;

    // workspace: XQ, XK, XV, OUTV  (4 x 40000 x 128 f32 = 82 MB)
    float* XQ   = (float*)d_ws;
    float* XK   = XQ + (size_t)N_PTS * DIM;
    float* XV   = XK + (size_t)N_PTS * DIM;
    float* OUTV = XV + (size_t)N_PTS * DIM;

    const dim3 blk(256);
    const int  gGemm = (N_PTS + DIM - 1) / DIM;     // 313
    gemm128_bias_kernel<<<gGemm, blk, 0, stream>>>(x, Wq, bq, XQ, N_PTS);
    gemm128_bias_kernel<<<gGemm, blk, 0, stream>>>(x, Wk, bk, XK, N_PTS);
    gemm128_bias_kernel<<<gGemm, blk, 0, stream>>>(x, Wv, bv, XV, N_PTS);
    ppf_attn_kernel<<<N_PTS / TILE_P, blk, 0, stream>>>(p, nrm, gidx, Wp1, bp1,
                                                        Wp2, bp2, XQ, XK, XV, OUTV);
    gemm128_bias_kernel<<<gGemm, blk, 0, stream>>>(OUTV, Wo, bo, out, N_PTS);
}
